// QuantTokenPredictor_29695403884916
// MI455X (gfx1250) — matrix operand order repaired and checked by reading, not compiled, not run on hardware
//
#include <hip/hip_runtime.h>
#include <hip/hip_bf16.h>

// ---------------------------------------------------------------------------
// Problem constants (from the reference): h is (4*2048)x256, w is 16384x256,
// out is 8192x16384 f32 distances.
// ---------------------------------------------------------------------------
#define MROWS 8192      // B*S
#define NCOLS 16384     // NUM_TOKS
#define KDIM  256       // TOK_DIM

typedef __attribute__((ext_vector_type(16))) __bf16 v16bf;
typedef __attribute__((ext_vector_type(8)))  float  v8f;

union FragU {
    uint4  u[2];   // two 16-byte global loads
    v16bf  v;      // 16 bf16 = 32 bytes, WMMA operand
};

// ---------------------------------------------------------------------------
// Kernel 1/2: f32 -> bf16 (round-to-nearest-even) + per-row squared L2 norm.
// One 256-thread block per row (D == 256).
// ---------------------------------------------------------------------------
__global__ __launch_bounds__(256)
void qtp_prep_bf16(const float* __restrict__ x,
                   unsigned short* __restrict__ xb,
                   float* __restrict__ x2)
{
    const int row = blockIdx.x;
    const int t   = threadIdx.x;
    const float v = x[(size_t)row * KDIM + t];

    // round-to-nearest-even f32 -> bf16
    unsigned int bits = __float_as_uint(v);
    unsigned int r    = bits + 0x7FFFu + ((bits >> 16) & 1u);
    xb[(size_t)row * KDIM + t] = (unsigned short)(r >> 16);

    // block reduction of v*v (8 waves of 32)
    float sq = v * v;
    #pragma unroll
    for (int off = 16; off > 0; off >>= 1)
        sq += __shfl_down(sq, off, 32);

    __shared__ float red[8];
    if ((t & 31) == 0) red[t >> 5] = sq;
    __syncthreads();
    if (t == 0) {
        float s = 0.f;
        #pragma unroll
        for (int i = 0; i < 8; ++i) s += red[i];
        x2[row] = s;
    }
}

// ---------------------------------------------------------------------------
// Kernel 3: WMMA GEMM cross = h_bf16 * w_bf16^T with fused distance epilogue.
// Block: 256 threads = 8 waves, tile 64(M) x 256(N). Waves laid out 2(M)x4(N),
// each wave computes a 32x64 patch = 2x4 tiles of 16x16 via
// v_wmma_f32_16x16x32_bf16, K = 256 in 8 chunks of 32.
// Both bf16 operand matrices (12 MB total) are L2-resident (192 MB), so
// fragment loads stream from L2; HBM traffic is dominated by the 512 MB
// f32 output -> kernel is store-bandwidth bound, which is the roofline floor.
// ---------------------------------------------------------------------------
__global__ __launch_bounds__(256)
void qtp_dist_wmma(const unsigned short* __restrict__ hb,   // MROWS x KDIM bf16
                   const unsigned short* __restrict__ wb,   // NCOLS x KDIM bf16
                   const float* __restrict__ h2,            // MROWS
                   const float* __restrict__ w2,            // NCOLS
                   float* __restrict__ out)                 // MROWS x NCOLS
{
    const int lane  = threadIdx.x & 31;
    const int wave  = threadIdx.x >> 5;
    const int waveM = wave >> 2;       // 0..1
    const int waveN = wave & 3;        // 0..3

    const int m0 = blockIdx.x * 64  + waveM * 32;   // wave M base (2 tiles)
    const int n0 = blockIdx.y * 256 + waveN * 64;   // wave N base (4 tiles)

    const int l16   = lane & 15;
    const int lhalf = lane >> 4;       // 0 or 1

    v8f acc[2][4];
    #pragma unroll
    for (int mi = 0; mi < 2; ++mi)
        #pragma unroll
        for (int ni = 0; ni < 4; ++ni)
            acc[mi][ni] = (v8f){0.f,0.f,0.f,0.f,0.f,0.f,0.f,0.f};

    // --- K loop: 8 chunks of 32 ---
    for (int k = 0; k < KDIM; k += 32) {
        // A fragments (16x32 bf16). Per ISA layout:
        //   lanes 0-15 : row = l16, K = {k..k+7} in V0-3, {k+16..k+23} in V4-7
        //   lanes 16-31: row = l16, K = {k+8..k+15},      {k+24..k+31}
        FragU a[2];
        #pragma unroll
        for (int mi = 0; mi < 2; ++mi) {
            const unsigned short* p =
                hb + (size_t)(m0 + mi * 16 + l16) * KDIM + k + lhalf * 8;
            a[mi].u[0] = *(const uint4*)(p);        // 8 bf16
            a[mi].u[1] = *(const uint4*)(p + 16);   // 8 bf16, +16 K elements
        }

        // B fragments (32x16 bf16, column n = l16). Per ISA layout:
        //   lanes 0-15 : K = {k..k+7}, {k+16..k+23}; lanes 16-31: K = {k+8..k+15}, {k+24..k+31}
        FragU b[4];
        #pragma unroll
        for (int ni = 0; ni < 4; ++ni) {
            const unsigned short* p =
                wb + (size_t)(n0 + ni * 16 + l16) * KDIM + k + lhalf * 8;
            b[ni].u[0] = *(const uint4*)(p);        // 8 bf16
            b[ni].u[1] = *(const uint4*)(p + 16);   // 8 bf16, +16 K elements
        }

        #pragma unroll
        for (int mi = 0; mi < 2; ++mi)
            #pragma unroll
            for (int ni = 0; ni < 4; ++ni)
                acc[mi][ni] = __builtin_amdgcn_wmma_f32_16x16x32_bf16(
                    /*neg_a=*/false, a[mi].v,
                    /*neg_b=*/false, b[ni].v,
                    /*c_mod=*/(short)0, acc[mi][ni],
                    /*reuse_a=*/false, /*reuse_b=*/false);
    }

    // --- Epilogue: d = sqrt(max(h2 - 2*cross + w2, 0)) ---
    // C/D layout: VGPR i, lane l -> row (8*lhalf + i), col (l&15).
    #pragma unroll
    for (int mi = 0; mi < 2; ++mi) {
        const int mrow = m0 + mi * 16 + lhalf * 8;  // rows mrow..mrow+7
        float4 h2lo = *(const float4*)(h2 + mrow);
        float4 h2hi = *(const float4*)(h2 + mrow + 4);
        float h2v[8] = {h2lo.x, h2lo.y, h2lo.z, h2lo.w,
                        h2hi.x, h2hi.y, h2hi.z, h2hi.w};
        #pragma unroll
        for (int ni = 0; ni < 4; ++ni) {
            const int ncol = n0 + ni * 16 + l16;
            const float w2s = w2[ncol];
            float* op = out + (size_t)mrow * NCOLS + ncol;
            #pragma unroll
            for (int i = 0; i < 8; ++i) {
                float sq = __builtin_fmaf(-2.0f, acc[mi][ni][i], h2v[i] + w2s);
                sq = fmaxf(sq, 0.0f);
                op[(size_t)i * NCOLS] = __builtin_sqrtf(sq);
            }
        }
    }
}

// ---------------------------------------------------------------------------
// Host-side launch. Workspace layout (needs ~12.1 MiB):
//   [0)            hb : MROWS*KDIM  ushort  (4 MiB)
//   [+4MiB)        wb : NCOLS*KDIM  ushort  (8 MiB)
//   [+12MiB)       h2 : MROWS       float
//   [+12MiB+32KiB) w2 : NCOLS       float
// ---------------------------------------------------------------------------
extern "C" void kernel_launch(void* const* d_in, const int* in_sizes, int n_in,
                              void* d_out, int out_size, void* d_ws, size_t ws_size,
                              hipStream_t stream)
{
    const float* h = (const float*)d_in[0];   // (B,S,D) f32
    const float* w = (const float*)d_in[1];   // (V,D)   f32

    unsigned short* hb = (unsigned short*)d_ws;
    unsigned short* wb = hb + (size_t)MROWS * KDIM;
    float* h2 = (float*)(wb + (size_t)NCOLS * KDIM);
    float* w2 = h2 + MROWS;

    qtp_prep_bf16<<<MROWS, 256, 0, stream>>>(h, hb, h2);
    qtp_prep_bf16<<<NCOLS, 256, 0, stream>>>(w, wb, w2);

    dim3 grid(MROWS / 64, NCOLS / 256);       // 128 x 64 blocks
    qtp_dist_wmma<<<grid, 256, 0, stream>>>(hb, wb, h2, w2, (float*)d_out);
}
